// RelativeMultiHeadAttentionBlock_25950192403066
// MI455X (gfx1250) — matrix operand order repaired and checked by reading, not compiled, not run on hardware
//
#include <hip/hip_runtime.h>
#include <hip/hip_bf16.h>

// ---------------------------------------------------------------------------
// Transformer-XL relative multi-head attention for MI455X (gfx1250, wave32).
// All GEMMs use v_wmma_f32_16x16x32_bf16. Attention is flash-fused so the
// [B,H,T,S] logits / relative bias never touch HBM (256MB each otherwise).
// K tiles are staged through LDS with the Tensor Data Mover (double buffered,
// s_wait_tensorcnt synchronized).
// ---------------------------------------------------------------------------

typedef __attribute__((ext_vector_type(16))) __bf16 bf16x16;
typedef __attribute__((ext_vector_type(8)))  __bf16 bf16x8;
typedef __attribute__((ext_vector_type(8)))  float  v8f;
typedef __attribute__((ext_vector_type(4)))  unsigned int u32x4;
typedef __attribute__((ext_vector_type(8)))  int  i32x8;
typedef __attribute__((ext_vector_type(4)))  int  i32x4;

constexpr int Bn  = 2;
constexpr int Tn  = 1024;
constexpr int Mn  = 1024;
constexpr int Fn  = 1024;
constexpr int Hn  = 16;
constexpr int HDn = 64;
constexpr int Sn  = Mn + Tn;              // 2048
constexpr float SCALE = 0.125f;           // 1/sqrt(64)
constexpr float NEGF  = -0.7f * 3.4028235e38f;

#define DEVFN static __device__ __forceinline__

DEVFN unsigned short f32_to_bf16_raw(float f) {
  union { float f; unsigned u; } v; v.f = f;
  unsigned u = v.u;
  u += 0x7FFFu + ((u >> 16) & 1u);        // round-to-nearest-even
  return (unsigned short)(u >> 16);
}

DEVFN v8f wmma_bf16(bf16x16 a, bf16x16 b, v8f c) {
  // (neg_a, A, neg_b, B, c_mod, C, reuse_a, reuse_b)
  return __builtin_amdgcn_wmma_f32_16x16x32_bf16(false, a, false, b,
                                                 (short)0, c, false, false);
}

// A fragment (16x32 bf16), row-major source, ld in elements.
// Lane layout (ISA 7.12.2): row = lane%16; lanes<16 hold K {0..7,16..23},
// lanes>=16 hold K {8..15,24..31}. Two 16-byte loads per lane.
DEVFN bf16x16 load_frag_a(const unsigned short* src, int ld) {
  const int lane = threadIdx.x & 31;
  const int row  = lane & 15;
  const int g    = (lane >> 4) * 8;
  const unsigned short* p = src + (size_t)row * ld + g;
  bf16x8 lo = *reinterpret_cast<const bf16x8*>(p);
  bf16x8 hi = *reinterpret_cast<const bf16x8*>(p + 16);
  bf16x16 a;
#pragma unroll
  for (int i = 0; i < 8; ++i) { a[i] = lo[i]; a[8 + i] = hi[i]; }
  return a;
}

// B fragment (32x16 bf16) from Bt stored N-major (row n holds its K values
// contiguously). Lane layout: col = lane%16; lanes<16 hold K {0..7,16..23},
// lanes>=16 hold K {8..15,24..31}. Two 16-byte loads per lane.
DEVFN bf16x16 load_frag_b(const unsigned short* srcT, int ld) {
  const int lane = threadIdx.x & 31;
  const int col  = lane & 15;
  const int kb   = (lane >> 4) * 8;
  const unsigned short* p = srcT + (size_t)col * ld + kb;
  bf16x8 lo = *reinterpret_cast<const bf16x8*>(p);
  bf16x8 hi = *reinterpret_cast<const bf16x8*>(p + 16);
  bf16x16 b;
#pragma unroll
  for (int i = 0; i < 8; ++i) { b[i] = lo[i]; b[8 + i] = hi[i]; }
  return b;
}

// ---------------------------------------------------------------------------
// Tensor Data Mover: contiguous `bytes8*8`-byte global -> LDS copy.
// D# per ISA cdna5 ch.8: group0 = {count=1, lds_addr, global_addr[56:0],
// type=2}; group1 = {mask=0, data_size=8B, tensor_dim0=tile_dim0=nelem,
// stride0=nelem, tensor_dim1=tile_dim1=1}. Groups 2/3 zero (<=2D tensor).
// ---------------------------------------------------------------------------
DEVFN void tdm_load_1d(const void* gsrc, unsigned lds_byte_off, unsigned nelem8) {
  unsigned long long ga = (unsigned long long)(uintptr_t)gsrc;
  u32x4 g0;
  g0[0] = 1u;                                        // count=1, no gather
  g0[1] = lds_byte_off;                              // lds_addr
  g0[2] = (unsigned)(ga & 0xFFFFFFFFu);              // global_addr[31:0]
  g0[3] = (unsigned)((ga >> 32) & 0x01FFFFFFu) | (2u << 30);  // [56:32] | type=2
  i32x8 g1;
  g1[0] = (int)(3u << 16);                           // data_size = 8 bytes
  g1[1] = (int)((nelem8 & 0xFFFFu) << 16);           // tensor_dim0[15:0]
  g1[2] = (int)((nelem8 >> 16) | (1u << 16));        // tensor_dim0[31:16], tensor_dim1=1
  g1[3] = (int)((nelem8 & 0xFFFFu) << 16);           // tile_dim0 = nelem8
  g1[4] = 1;                                         // tile_dim1 = 1, tile_dim2 = 0
  g1[5] = (int)nelem8;                               // tensor_dim0_stride[31:0]
  g1[6] = 0;
  g1[7] = 0;
  i32x4 gz = {0, 0, 0, 0};
#if __clang_major__ >= 23
  i32x8 gz8 = {0, 0, 0, 0, 0, 0, 0, 0};
  __builtin_amdgcn_tensor_load_to_lds(g0, g1, gz, gz, gz8, 0);
#else
  __builtin_amdgcn_tensor_load_to_lds(g0, g1, gz, gz, 0);
#endif
}

// ---------------------------------------------------------------------------
// Packing / elementwise prep kernels
// ---------------------------------------------------------------------------

// kv_in = concat(memory, x) -> bf16; also emits x in bf16.
__global__ void pack_kv_kernel(const float* __restrict__ x,
                               const float* __restrict__ mem,
                               unsigned short* __restrict__ kvbf,
                               unsigned short* __restrict__ xbf) {
  size_t i = (size_t)blockIdx.x * blockDim.x + threadIdx.x;
  if (i >= (size_t)Bn * Sn * Fn) return;
  int f = (int)(i % Fn);
  size_t rs = i / Fn;
  int s = (int)(rs % Sn);
  int b = (int)(rs / Sn);
  float v;
  if (s < Mn) {
    v = mem[((size_t)b * Mn + s) * Fn + f];
  } else {
    v = x[((size_t)b * Tn + (s - Mn)) * Fn + f];
    xbf[((size_t)b * Tn + (s - Mn)) * Fn + f] = f32_to_bf16_raw(v);
  }
  kvbf[i] = f32_to_bf16_raw(v);
}

// dst[c*rows + r] = bf16(src[r*cols + c])
__global__ void transpose_to_bf16_kernel(const float* __restrict__ src,
                                         unsigned short* __restrict__ dst,
                                         int rows, int cols) {
  size_t i = (size_t)blockIdx.x * blockDim.x + threadIdx.x;
  if (i >= (size_t)rows * cols) return;
  int c = (int)(i % cols);
  int r = (int)(i / cols);
  dst[(size_t)c * rows + r] = f32_to_bf16_raw(src[i]);
}

// Transformer-XL sinusoidal table for pos = S-1..0, [S, F] bf16.
__global__ void sinus_kernel(unsigned short* __restrict__ s) {
  size_t i = (size_t)blockIdx.x * blockDim.x + threadIdx.x;
  if (i >= (size_t)Sn * Fn) return;
  int f = (int)(i % Fn);
  int m = (int)(i / Fn);
  float pos = (float)(Sn - 1 - m);
  int idx = (f < Fn / 2) ? f : f - Fn / 2;
  // inv_freq = 10000^(-idx/512) = exp(-idx * ln(1e4)/512)
  float invf = __expf(-(float)idx * (9.210340371976184f / 512.0f));
  float ang = pos * invf;
  s[i] = f32_to_bf16_raw((f < Fn / 2) ? __sinf(ang) : __cosf(ang));
}

__global__ void zero_bf16_kernel(unsigned short* __restrict__ p, size_t n) {
  size_t i = (size_t)blockIdx.x * blockDim.x + threadIdx.x;
  if (i < n) p[i] = 0;
}

// Reverse inclusive cumsums used by the Transformer-XL segment mask.
__global__ void cumsum_kernel(const int* __restrict__ mask,
                              const int* __restrict__ mmask,
                              int* __restrict__ qcum,
                              int* __restrict__ kcum) {
  int b = blockIdx.x;
  if (threadIdx.x != 0) return;
  int acc = 0;
  for (int t = Tn - 1; t >= 0; --t) { acc += mask[b * Tn + t]; qcum[b * Tn + t] = acc; }
  acc = 0;
  for (int s = Sn - 1; s >= 0; --s) {
    int mv = (s >= Mn) ? mask[b * Tn + (s - Mn)] : mmask[b * Mn + s];
    acc += mv;
    kcum[b * Sn + s] = acc;
  }
}

// ---------------------------------------------------------------------------
// WMMA GEMM, register-blocked 32x64 per wave:
//   C tile = A[M,K](bf16, row-major) x Bt[N,K](bf16, N-major)^T
// 2 A-fragments x 4 B-fragments -> 8 accumulators, 8 WMMAs per 32-wide
// k-step. mode selects the epilogue / output layout.
// ---------------------------------------------------------------------------
enum { MODE_QUQV = 0, MODE_K = 1, MODE_VT = 2, MODE_R = 3, MODE_OUT = 4 };

__global__ __launch_bounds__(32)
void wmma_gemm_kernel(const unsigned short* __restrict__ A,
                      const unsigned short* __restrict__ Bt,
                      int K, int mode,
                      const float* __restrict__ bias0,
                      const float* __restrict__ bias1,
                      unsigned short* __restrict__ out0,
                      unsigned short* __restrict__ out1,
                      float* __restrict__ outf) {
  const int lane = threadIdx.x & 31;
  const int col0 = blockIdx.x * 64;
  const int row0 = blockIdx.y * 32;
  const unsigned short* ap0 = A + (size_t)row0 * K;
  const unsigned short* ap1 = A + (size_t)(row0 + 16) * K;
  v8f acc[2][4];
#pragma unroll
  for (int rt = 0; rt < 2; ++rt)
#pragma unroll
    for (int ct = 0; ct < 4; ++ct)
      acc[rt][ct] = (v8f){0.f,0.f,0.f,0.f,0.f,0.f,0.f,0.f};

  for (int kk = 0; kk < K; kk += 32) {
    __builtin_prefetch(ap0 + kk + 256, 0, 0);        // global_prefetch_b8
    __builtin_prefetch(ap1 + kk + 256, 0, 0);
    bf16x16 a0 = load_frag_a(ap0 + kk, K);
    bf16x16 a1 = load_frag_a(ap1 + kk, K);
#pragma unroll
    for (int ct = 0; ct < 4; ++ct) {
      bf16x16 b = load_frag_b(Bt + (size_t)(col0 + ct * 16) * K + kk, K);
      acc[0][ct] = wmma_bf16(a0, b, acc[0][ct]);
      acc[1][ct] = wmma_bf16(a1, b, acc[1][ct]);
    }
  }

  const int hh = lane >> 4;
#pragma unroll
  for (int rt = 0; rt < 2; ++rt) {
#pragma unroll
    for (int ct = 0; ct < 4; ++ct) {
      const int n = col0 + ct * 16 + (lane & 15);
#pragma unroll
      for (int e = 0; e < 8; ++e) {
        int r = row0 + rt * 16 + e + 8 * hh;   // C layout: row = v + 8*(lane/16)
        float v = acc[rt][ct][e];
        if (mode == MODE_QUQV) {
          int b = r >> 10, t = r & 1023;       // rows span B*T
          int h = n >> 6,  d = n & 63;
          size_t o = ((size_t)(b * Hn + h) * Tn + t) * HDn + d;
          out0[o] = f32_to_bf16_raw(v + bias0[n]);   // q + u_bias
          out1[o] = f32_to_bf16_raw(v + bias1[n]);   // q + v_bias
        } else if (mode == MODE_K) {
          int b = r >> 11, s = r & 2047;       // rows span B*S
          int h = n >> 6,  d = n & 63;
          out0[((size_t)(b * Hn + h) * Sn + s) * HDn + d] = f32_to_bf16_raw(v);
        } else if (mode == MODE_VT) {
          int b = r >> 11, s = r & 2047;
          int h = n >> 6,  d = n & 63;
          out0[((size_t)(b * Hn + h) * HDn + d) * Sn + s] = f32_to_bf16_raw(v);
        } else if (mode == MODE_R) {
          out0[(size_t)r * (Hn * HDn) + n] = f32_to_bf16_raw(v);
        } else { // MODE_OUT
          outf[(size_t)r * Fn + n] = v + bias0[n];
        }
      }
    }
  }
}

// ---------------------------------------------------------------------------
// Flash attention with relative-shift bias. One wave handles a 16-row query
// tile for one (b,h); iterates S in steps of 32. Per step: 4 WMMAs (QK^T,
// K tile TDM-staged through LDS double buffer), 6 WMMAs (bias band, width 48,
// gathered diagonally from LDS), online softmax with wave shuffles, 4 WMMAs
// (P*V with vT for contiguous K-dim loads).
// ---------------------------------------------------------------------------
__global__ __launch_bounds__(32)
void attn_flash_kernel(const unsigned short* __restrict__ qu,
                       const unsigned short* __restrict__ qv,
                       const unsigned short* __restrict__ kbuf,
                       const unsigned short* __restrict__ vT,
                       const unsigned short* __restrict__ rbuf,
                       const int* __restrict__ qcum,
                       const int* __restrict__ kcum,
                       unsigned short* __restrict__ attnout) {
  __shared__ __align__(16) unsigned short kTile[2][32 * HDn];  // 2 x 4KB
  __shared__ __align__(16) float bandP[16][48];
  __shared__ __align__(16) unsigned short probsS[16][32];

  const int lane = threadIdx.x & 31;
  const int hh = lane >> 4;
  const int cl = lane & 15;
  const int t0 = blockIdx.x * 16;
  const int h  = blockIdx.y;
  const int b  = blockIdx.z;

  const unsigned short* quB = qu + ((size_t)(b * Hn + h) * Tn + t0) * HDn;
  const unsigned short* qvB = qv + ((size_t)(b * Hn + h) * Tn + t0) * HDn;
  bf16x16 aqu[2], aqv[2];
  aqu[0] = load_frag_a(quB, HDn);  aqu[1] = load_frag_a(quB + 32, HDn);
  aqv[0] = load_frag_a(qvB, HDn);  aqv[1] = load_frag_a(qvB + 32, HDn);

  float Mrow[8], Lrow[8];
  int qc[8];
  v8f O[4];
#pragma unroll
  for (int e = 0; e < 8; ++e) {
    Mrow[e] = -3.0e38f;
    Lrow[e] = 0.f;
    qc[e] = qcum[b * Tn + t0 + e + 8 * hh];
  }
#pragma unroll
  for (int j = 0; j < 4; ++j) O[j] = (v8f){0.f,0.f,0.f,0.f,0.f,0.f,0.f,0.f};

  const unsigned short* kB = kbuf + (size_t)(b * Hn + h) * Sn * HDn;
  const unsigned short* vB = vT   + (size_t)(b * Hn + h) * HDn * Sn;

  // TDM prologue: stage first K tile (32 rows x 64 bf16 = 4KB, contiguous).
  int cur = 0;
  tdm_load_1d(kB, (unsigned)(uintptr_t)&kTile[0][0], (32 * HDn * 2) / 8);

  for (int s0 = 0; s0 < Sn; s0 += 32) {
    const bool havenext = (s0 + 32 < Sn);
    if (havenext) {   // prefetch next K tile into the other buffer
      tdm_load_1d(kB + (size_t)(s0 + 32) * HDn,
                  (unsigned)(uintptr_t)&kTile[cur ^ 1][0], (32 * HDn * 2) / 8);
      __builtin_amdgcn_s_wait_tensorcnt(1);   // current tile complete
    } else {
      __builtin_amdgcn_s_wait_tensorcnt(0);
    }
    asm volatile("" ::: "memory");            // keep LDS reads below the wait

    // ---- logits (QK^T), two 16x16 tiles, K=64, K-operand from LDS ----
    v8f c[2];
#pragma unroll
    for (int half = 0; half < 2; ++half) {
      v8f cc = (v8f){0.f,0.f,0.f,0.f,0.f,0.f,0.f,0.f};
      const unsigned short* kp = &kTile[cur][(size_t)(16 * half) * HDn];
      cc = wmma_bf16(aqu[0], load_frag_b(kp,      HDn), cc);
      cc = wmma_bf16(aqu[1], load_frag_b(kp + 32, HDn), cc);
      c[half] = cc;
    }

    // ---- relative bias band: P[w,p] = (q+v_bias)[t0+w] . r[mb+p], p<48 ----
    // bias[t,s] = P[t-t0, 15 + (s-s0) - (t-t0)], mb = s0-t0+T-16 (>=0, in r).
    const int mb = s0 - t0 + Tn - 16;
#pragma unroll
    for (int nb = 0; nb < 3; ++nb) {
      v8f cb = (v8f){0.f,0.f,0.f,0.f,0.f,0.f,0.f,0.f};
      const unsigned short* rp = rbuf + (size_t)(mb + nb * 16) * (Hn * HDn) + h * HDn;
      cb = wmma_bf16(aqv[0], load_frag_b(rp,      Hn * HDn), cb);
      cb = wmma_bf16(aqv[1], load_frag_b(rp + 32, Hn * HDn), cb);
#pragma unroll
      for (int e = 0; e < 8; ++e) bandP[e + 8 * hh][nb * 16 + cl] = cb[e];
    }
    __syncthreads();

    // ---- mask + scale + bias, running max ----
    const int kc0 = kcum[b * Sn + s0 + cl];
    const int kc1 = kcum[b * Sn + s0 + 16 + cl];
    float nm[8];
#pragma unroll
    for (int e = 0; e < 8; ++e) {
      const int row = e + 8 * hh;
      const int t = t0 + row;
      {
        const int s = s0 + cl;
        float l = (c[0][e] + bandP[row][15 + cl - row]) * SCALE;
        bool ok = (qc[e] == kc0) && (s <= t);
        c[0][e] = ok ? l : NEGF;
      }
      {
        const int s = s0 + 16 + cl;
        float l = (c[1][e] + bandP[row][31 + cl - row]) * SCALE;
        bool ok = (qc[e] == kc1) && (s <= t);
        c[1][e] = ok ? l : NEGF;
      }
      float m = fmaxf(c[0][e], c[1][e]);
#pragma unroll
      for (int x = 1; x < 16; x <<= 1) m = fmaxf(m, __shfl_xor(m, x, 32));
      nm[e] = fmaxf(Mrow[e], m);
    }

    // ---- probs, online rescale, stage probs to LDS as bf16 A operand ----
#pragma unroll
    for (int e = 0; e < 8; ++e) {
      const float alpha = __expf(Mrow[e] - nm[e]);
      Mrow[e] = nm[e];
      const float p0 = __expf(c[0][e] - nm[e]);
      const float p1 = __expf(c[1][e] - nm[e]);
      const int row = e + 8 * hh;
      probsS[row][cl]      = f32_to_bf16_raw(p0);
      probsS[row][16 + cl] = f32_to_bf16_raw(p1);
      float rs = p0 + p1;
#pragma unroll
      for (int x = 1; x < 16; x <<= 1) rs += __shfl_xor(rs, x, 32);
      Lrow[e] = Lrow[e] * alpha + rs;
#pragma unroll
      for (int j = 0; j < 4; ++j) O[j][e] *= alpha;
    }
    __syncthreads();

    // ---- O += P(16x32) x V(32x64) ----
    bf16x16 pa = load_frag_a(&probsS[0][0], 32);
#pragma unroll
    for (int j = 0; j < 4; ++j) {
      const unsigned short* vp = vB + (size_t)(j * 16) * Sn + s0;
      O[j] = wmma_bf16(pa, load_frag_b(vp, Sn), O[j]);
    }
    __syncthreads();
    cur ^= 1;
  }

  // ---- normalize and store attn[b,t,h,d] (row-major for output GEMM) ----
#pragma unroll
  for (int e = 0; e < 8; ++e) {
    const int t = t0 + e + 8 * hh;
    const float inv = 1.0f / Lrow[e];
#pragma unroll
    for (int j = 0; j < 4; ++j) {
      const int d = j * 16 + cl;
      attnout[((size_t)(b * Tn + t) * Hn + h) * HDn + d] =
          f32_to_bf16_raw(O[j][e] * inv);
    }
  }
}

// ---------------------------------------------------------------------------
// Host launcher
// ---------------------------------------------------------------------------
extern "C" void kernel_launch(void* const* d_in, const int* in_sizes, int n_in,
                              void* d_out, int out_size, void* d_ws, size_t ws_size,
                              hipStream_t stream) {
  (void)in_sizes; (void)n_in; (void)out_size; (void)ws_size;
  const float* x     = (const float*)d_in[0];
  const float* mem   = (const float*)d_in[1];
  const int*   mask  = (const int*)d_in[2];
  const int*   mmask = (const int*)d_in[3];
  const float* Wq    = (const float*)d_in[4];
  const float* Wk    = (const float*)d_in[5];
  const float* Wv    = (const float*)d_in[6];
  const float* Wr    = (const float*)d_in[7];
  const float* ub    = (const float*)d_in[8];
  const float* vb    = (const float*)d_in[9];
  const float* Wo    = (const float*)d_in[10];
  const float* bo    = (const float*)d_in[11];
  float* out = (float*)d_out;

  char* w = (char*)d_ws;
  auto alloc = [&](size_t bytes) -> char* {
    char* p = w;
    w += (bytes + 255) & ~(size_t)255;
    return p;
  };
  unsigned short* xbf   = (unsigned short*)alloc((size_t)Bn * Tn * Fn * 2);
  unsigned short* kvbf  = (unsigned short*)alloc((size_t)Bn * Sn * Fn * 2);
  unsigned short* WqT   = (unsigned short*)alloc((size_t)Fn * Fn * 2);
  unsigned short* WkT   = (unsigned short*)alloc((size_t)Fn * Fn * 2);
  unsigned short* WvT   = (unsigned short*)alloc((size_t)Fn * Fn * 2);
  unsigned short* WrT   = (unsigned short*)alloc((size_t)Fn * Fn * 2);
  unsigned short* WoT   = (unsigned short*)alloc((size_t)Fn * Fn * 2);
  unsigned short* sinus = (unsigned short*)alloc((size_t)Sn * Fn * 2);
  unsigned short* qubuf = (unsigned short*)alloc((size_t)Bn * Hn * Tn * HDn * 2);
  unsigned short* qvbuf = (unsigned short*)alloc((size_t)Bn * Hn * Tn * HDn * 2);
  unsigned short* kbuf  = (unsigned short*)alloc((size_t)Bn * Hn * Sn * HDn * 2);
  unsigned short* vTbuf = (unsigned short*)alloc((size_t)Bn * Hn * HDn * Sn * 2);
  unsigned short* rbuf  = (unsigned short*)alloc((size_t)(Sn + Tn) * Fn * 2);
  unsigned short* attnb = (unsigned short*)alloc((size_t)Bn * Tn * Fn * 2);
  int* qcum = (int*)alloc((size_t)Bn * Tn * 4);
  int* kcum = (int*)alloc((size_t)Bn * Sn * 4);

  // --- packing / prep ---
  {
    size_t n = (size_t)Bn * Sn * Fn;
    pack_kv_kernel<<<dim3((unsigned)((n + 255) / 256)), 256, 0, stream>>>(x, mem, kvbf, xbf);
  }
  {
    size_t n = (size_t)Fn * Fn;
    dim3 g((unsigned)((n + 255) / 256));
    transpose_to_bf16_kernel<<<g, 256, 0, stream>>>(Wq, WqT, Fn, Fn);
    transpose_to_bf16_kernel<<<g, 256, 0, stream>>>(Wk, WkT, Fn, Fn);
    transpose_to_bf16_kernel<<<g, 256, 0, stream>>>(Wv, WvT, Fn, Fn);
    transpose_to_bf16_kernel<<<g, 256, 0, stream>>>(Wr, WrT, Fn, Fn);
    transpose_to_bf16_kernel<<<g, 256, 0, stream>>>(Wo, WoT, Fn, Fn); // [H*HD,F] -> [F,H*HD]
  }
  {
    size_t n = (size_t)Sn * Fn;
    sinus_kernel<<<dim3((unsigned)((n + 255) / 256)), 256, 0, stream>>>(sinus);
    size_t z = (size_t)Tn * Fn; // zero r tail rows [S, S+T)
    zero_bf16_kernel<<<dim3((unsigned)((z + 255) / 256)), 256, 0, stream>>>(rbuf + (size_t)Sn * Fn, z);
  }
  cumsum_kernel<<<dim3(Bn), 32, 0, stream>>>(mask, mmask, qcum, kcum);

  // --- projections (WMMA, 32x64 register-blocked) ---
  wmma_gemm_kernel<<<dim3(Fn / 64, (Bn * Tn) / 32), 32, 0, stream>>>(
      xbf, WqT, Fn, MODE_QUQV, ub, vb, qubuf, qvbuf, nullptr);
  wmma_gemm_kernel<<<dim3(Fn / 64, (Bn * Sn) / 32), 32, 0, stream>>>(
      kvbf, WkT, Fn, MODE_K, nullptr, nullptr, kbuf, nullptr, nullptr);
  wmma_gemm_kernel<<<dim3(Fn / 64, (Bn * Sn) / 32), 32, 0, stream>>>(
      kvbf, WvT, Fn, MODE_VT, nullptr, nullptr, vTbuf, nullptr, nullptr);
  wmma_gemm_kernel<<<dim3(Fn / 64, Sn / 32), 32, 0, stream>>>(
      sinus, WrT, Fn, MODE_R, nullptr, nullptr, rbuf, nullptr, nullptr);

  // --- fused flash attention with relative-shift bias (WMMA + TDM) ---
  attn_flash_kernel<<<dim3(Tn / 16, Hn, Bn), 32, 0, stream>>>(
      qubuf, qvbuf, kbuf, vTbuf, rbuf, qcum, kcum, attnb);

  // --- output projection (WMMA) ---
  wmma_gemm_kernel<<<dim3(Fn / 64, (Bn * Tn) / 32), 32, 0, stream>>>(
      attnb, WoT, Fn, MODE_OUT, bo, nullptr, nullptr, nullptr, out);
}
